// PAM_Module_37606733644460
// MI455X (gfx1250) — matrix operand order repaired and checked by reading, compiled, not run on hardware
//
#include <hip/hip_runtime.h>

typedef _Float16 v8h  __attribute__((ext_vector_type(8)));
typedef _Float16 v16h __attribute__((ext_vector_type(16)));
typedef float    v8f  __attribute__((ext_vector_type(8)));

#define NB 4
#define NC 512
#define NC8 64
#define NN 4096

static __device__ __forceinline__ v16h cat16(v8h lo, v8h hi) {
  return __builtin_shufflevector(lo, hi, 0,1,2,3,4,5,6,7,8,9,10,11,12,13,14,15);
}
static __device__ __forceinline__ v8f wmma16(v16h a, v16h b, v8f c) {
  // D = A(16x32 f16) * B(32x16 f16) + C(16x16 f32)
  return __builtin_amdgcn_wmma_f32_16x16x32_f16(false, a, false, b, (short)0, c, false, false);
}

#if __has_builtin(__builtin_amdgcn_tensor_load_to_lds) && __has_builtin(__builtin_amdgcn_s_wait_tensorcnt)
#define USE_TDM 1
typedef unsigned int u32x4 __attribute__((ext_vector_type(4)));
typedef int          i32x4 __attribute__((ext_vector_type(4)));
typedef int          i32x8 __attribute__((ext_vector_type(8)));

// DMA one V tile (512 rows x 32 f16, row stride 4096 elems) into LDS at lds_off.
// D# packed per CDNA5 ISA 8.3/8.4 (2-D tensor, groups 2/3 zero).
static __device__ __forceinline__ void tdm_load_vtile(const _Float16* src, unsigned lds_off) {
  unsigned long long ga = (unsigned long long)(uintptr_t)src;
  u32x4 g0;
  g0[0] = 1u;                                               // count=1, user descriptor
  g0[1] = lds_off;                                          // lds_addr (bytes)
  g0[2] = (unsigned)ga;                                     // global_addr[31:0]
  g0[3] = ((unsigned)(ga >> 32) & 0x01FFFFFFu) | (2u << 30); // addr[56:32] | type=2
  i32x8 g1;
  g1[0] = (int)(1u << 16);                 // workgroup_mask=0, data_size=1 (2 bytes)
  g1[1] = (int)(4096u << 16);              // tensor_dim0 = 4096 (bits 79:48, low 16)
  g1[2] = (int)(512u << 16);               // tensor_dim0 hi=0 | tensor_dim1 = 512
  g1[3] = (int)(32u << 16);                // tensor_dim1 hi=0 | tile_dim0 = 32
  g1[4] = (int)(512u);                     // tile_dim1 = 512, tile_dim2 = 0
  g1[5] = (int)(4096u);                    // tensor_dim0_stride = 4096 (low 32)
  g1[6] = 0;                               // stride hi | tensor_dim1_stride (unused 2D)
  g1[7] = 0;
  i32x4 z4 = {0, 0, 0, 0};
  i32x8 z8 = {0, 0, 0, 0, 0, 0, 0, 0};
  __builtin_amdgcn_tensor_load_to_lds(g0, g1, z4, z4, z8, 0);
}
#endif

// ---------------------------------------------------------------------------
// Kernel 1: projections.  OUT^T[o,n] = sum_c W[o,c] * x[b,c,n] + bias[o]
// q,k stored [b][n][64] f16 ; v stored [b][c][4096] f16
// One wave computes a 64(o) x 16(n) tile.  Jobs: q:1024, k:1024, v:8192.
// ---------------------------------------------------------------------------
__global__ __launch_bounds__(256) void pam_proj_kernel(
    const float* __restrict__ x,
    const float* __restrict__ wq, const float* __restrict__ bq,
    const float* __restrict__ wk, const float* __restrict__ bk,
    const float* __restrict__ wv, const float* __restrict__ bv,
    _Float16* __restrict__ qws, _Float16* __restrict__ kws,
    _Float16* __restrict__ vws)
{
  const int job  = blockIdx.x * 8 + (threadIdx.x >> 5);
  const int lane = threadIdx.x & 31;
  const int hf   = (lane < 16) ? 0 : 1;   // half-wave id
  const int l15  = lane & 15;

  int kind, b, nt, og;
  const float* W; const float* bias;
  if (job < 1024)      { kind = 0; b = job >> 8;          nt = job & 255;  og = 0;            W = wq; bias = bq; }
  else if (job < 2048) { int j = job - 1024; kind = 1; b = j >> 8; nt = j & 255; og = 0;      W = wk; bias = bk; }
  else                 { int j = job - 2048; kind = 2; b = j >> 11; og = (j >> 8) & 7; nt = j & 255; W = wv; bias = bv; }

  const int o0 = og * 64;
  const int n  = nt * 16 + l15;
  const float* xb = x + (size_t)b * NC * NN;

  v8f acc[4] = {};

  for (int cc = 0; cc < NC; cc += 32) {
    // B operand: X chunk (32 c-dims x 16 n).  lane=n, K-pairs of c.
    v16h Bx;
    const int cb = cc + hf * 16;
    #pragma unroll
    for (int v = 0; v < 8; v++) {
      float f0 = xb[(size_t)(cb + 2 * v)     * NN + n];
      float f1 = xb[(size_t)(cb + 2 * v + 1) * NN + n];
      Bx[2 * v]     = (_Float16)f0;
      Bx[2 * v + 1] = (_Float16)f1;
    }
    #pragma unroll
    for (int mt = 0; mt < 4; mt++) {
      // A operand: W rows (16 o x 32 c).  lane=o row, 16 contiguous c.
      const int o = o0 + mt * 16 + l15;
      const float* wp = W + (size_t)o * NC + cc + hf * 16;
      float4 w0 = *(const float4*)(wp);
      float4 w1 = *(const float4*)(wp + 4);
      float4 w2 = *(const float4*)(wp + 8);
      float4 w3 = *(const float4*)(wp + 12);
      v16h A;
      A[0]=(_Float16)w0.x;  A[1]=(_Float16)w0.y;  A[2]=(_Float16)w0.z;  A[3]=(_Float16)w0.w;
      A[4]=(_Float16)w1.x;  A[5]=(_Float16)w1.y;  A[6]=(_Float16)w1.z;  A[7]=(_Float16)w1.w;
      A[8]=(_Float16)w2.x;  A[9]=(_Float16)w2.y;  A[10]=(_Float16)w2.z; A[11]=(_Float16)w2.w;
      A[12]=(_Float16)w3.x; A[13]=(_Float16)w3.y; A[14]=(_Float16)w3.z; A[15]=(_Float16)w3.w;
      acc[mt] = wmma16(A, Bx, acc[mt]);
    }
  }

  // bias + store (C layout: VGPR r -> rows r / r+8 by half-wave, lane -> n)
  #pragma unroll
  for (int mt = 0; mt < 4; mt++) {
    #pragma unroll
    for (int r = 0; r < 8; r++) {
      const int o = o0 + mt * 16 + r + hf * 8;
      float val = acc[mt][r] + bias[o];
      _Float16 h = (_Float16)val;
      if (kind == 2) {
        vws[((size_t)b * NC + o) * NN + n] = h;
      } else {
        _Float16* d = (kind == 0) ? qws : kws;
        d[((size_t)b * NN + n) * NC8 + o] = h;
      }
    }
  }
}

// ---------------------------------------------------------------------------
// Kernel 2: flash-style attention + epilogue.
// Workgroup: 64 query rows of one batch, 8 waves, 256 threads.
// Pass 1: row max (waves split j).  Pass 2: wave(mg,cg) rows 16, channels 256.
// V tiles staged into LDS by the Tensor Data Mover, double buffered.
// Dynamic LDS layout:
//   [0, 64K)      : ldsV, two 32KB V tiles [c=512][j=32] f16
//   [64K, 72K)    : ldsP, 8 x 1KB per-wave scratch (P tile / f32 transpose)
//   [72K, 74K)    : smax, 8 x 64 f32 pass-1 partial row maxima
// ---------------------------------------------------------------------------
__global__ __launch_bounds__(256) void pam_attn_kernel(
    const float* __restrict__ x, const float* __restrict__ gammap,
    const _Float16* __restrict__ qws, const _Float16* __restrict__ kws,
    const _Float16* __restrict__ vws, float* __restrict__ out)
{
  extern __shared__ char smem[];
  _Float16* ldsV = (_Float16*)smem;                    // 2 * NC * 32 halves
  _Float16* ldsP = (_Float16*)(smem + 65536);          // 8 * 512 halves
  float*    smax = (float*)(smem + 65536 + 8192);      // 8 * 64 floats

  const int wg   = blockIdx.x;
  const int b    = wg >> 6;
  const int i0   = (wg & 63) << 6;
  const int tid  = threadIdx.x;
  const int wave = tid >> 5;
  const int lane = tid & 31;
  const int hf   = (lane < 16) ? 0 : 1;
  const int l15  = lane & 15;

  const _Float16* qb = qws + (size_t)b * NN * NC8;
  const _Float16* kb = kws + (size_t)b * NN * NC8;
  const _Float16* vb = vws + (size_t)b * NC * NN;

#if USE_TDM
  const unsigned ldsV_off0 = (unsigned)(uintptr_t)(void*)ldsV;
  const unsigned ldsV_off1 = ldsV_off0 + (unsigned)(NC * 32 * sizeof(_Float16));
  // Prologue: start DMA of V tile 0 while pass 1 computes.
  if (wave == 0) tdm_load_vtile(vb, ldsV_off0);
#endif

  // ---------------- Pass 1: row maxima, waves split the j range -------------
  v16h Qa[4][2];
  #pragma unroll
  for (int mt = 0; mt < 4; mt++) {
    const int n = i0 + mt * 16 + l15;
    const _Float16* qp = qb + (size_t)n * NC8;
    #pragma unroll
    for (int ch = 0; ch < 2; ch++) {
      v8h lo = *(const v8h*)(qp + ch * 32 + hf * 16);
      v8h hi = *(const v8h*)(qp + ch * 32 + 8 + hf * 16);
      Qa[mt][ch] = cat16(lo, hi);
    }
  }

  float pmax[4][8];
  #pragma unroll
  for (int mt = 0; mt < 4; mt++)
    #pragma unroll
    for (int r = 0; r < 8; r++) pmax[mt][r] = -1e30f;

  for (int tt = 0; tt < 16; tt++) {
    const int j0 = (wave + tt * 8) * 32;
    v16h Kb[2][2];
    #pragma unroll
    for (int nt = 0; nt < 2; nt++) {
      const int j = j0 + nt * 16 + l15;
      #pragma unroll
      for (int ch = 0; ch < 2; ch++) {
        const _Float16* kp = kb + (size_t)j * NC8 + ch * 32 + hf * 16;
        Kb[nt][ch] = cat16(*(const v8h*)(kp), *(const v8h*)(kp + 8));
      }
    }
    #pragma unroll
    for (int mt = 0; mt < 4; mt++) {
      #pragma unroll
      for (int nt = 0; nt < 2; nt++) {
        v8f z = {};
        v8f e = wmma16(Qa[mt][1], Kb[nt][1], z);
        e = wmma16(Qa[mt][0], Kb[nt][0], e);
        #pragma unroll
        for (int r = 0; r < 8; r++) pmax[mt][r] = fmaxf(pmax[mt][r], e[r]);
      }
    }
  }
  // reduce across the 16 lanes of each half-wave
  #pragma unroll
  for (int mt = 0; mt < 4; mt++)
    #pragma unroll
    for (int r = 0; r < 8; r++) {
      float m = pmax[mt][r];
      m = fmaxf(m, __shfl_xor(m, 1, 32));
      m = fmaxf(m, __shfl_xor(m, 2, 32));
      m = fmaxf(m, __shfl_xor(m, 4, 32));
      m = fmaxf(m, __shfl_xor(m, 8, 32));
      pmax[mt][r] = m;
    }
  if (l15 == 0) {
    #pragma unroll
    for (int mt = 0; mt < 4; mt++)
      #pragma unroll
      for (int r = 0; r < 8; r++)
        smax[wave * 64 + mt * 16 + r + hf * 8] = pmax[mt][r];
  }
  __syncthreads();

  // ---------------- Pass 2: wave = (row group mg, channel group cg) ---------
  const int mg = wave >> 1;
  const int cg = wave & 1;
  const int c0 = cg * 256;

  float rowmax[8];
  #pragma unroll
  for (int r = 0; r < 8; r++) {
    float m = -1e30f;
    const int row = mg * 16 + r + hf * 8;
    #pragma unroll
    for (int w2 = 0; w2 < 8; w2++) m = fmaxf(m, smax[w2 * 64 + row]);
    rowmax[r] = m;
  }

  v16h Qm[2];
  {
    const int n = i0 + mg * 16 + l15;
    const _Float16* qp = qb + (size_t)n * NC8;
    #pragma unroll
    for (int ch = 0; ch < 2; ch++) {
      v8h lo = *(const v8h*)(qp + ch * 32 + hf * 16);
      v8h hi = *(const v8h*)(qp + ch * 32 + 8 + hf * 16);
      Qm[ch] = cat16(lo, hi);
    }
  }

  v8f acc[16] = {};
  float lsum[8];
  #pragma unroll
  for (int r = 0; r < 8; r++) lsum[r] = 0.0f;

  _Float16* pp = ldsP + wave * 512;

  for (int t = 0; t < 128; t++) {
    const int j0 = t * 32;
#if USE_TDM
    // Tile t was DMA'd by wave 0; wait, publish via barrier, then prefetch t+1
    // into the other buffer (barrier guarantees everyone is done reading it).
    if (wave == 0) __builtin_amdgcn_s_wait_tensorcnt(0);
    __syncthreads();
    if (wave == 0 && (t + 1) < 128)
      tdm_load_vtile(vb + (size_t)(t + 1) * 32, ((t + 1) & 1) ? ldsV_off1 : ldsV_off0);
#else
    __syncthreads();
    // cooperative stage of V tile [512 c][32 j] into LDS
    {
      _Float16* dst = ldsV + (t & 1) * (NC * 32);
      #pragma unroll
      for (int kk = 0; kk < 8; kk++) {
        const int id = tid + kk * 256;
        const int c  = id >> 2;
        const int qd = id & 3;
        float4 d = *(const float4*)(vb + (size_t)c * NN + j0 + qd * 8);
        *(float4*)(&dst[c * 32 + qd * 8]) = d;
      }
    }
    __syncthreads();
#endif
    const _Float16* ldsVcur = ldsV + (t & 1) * (NC * 32);

    // E = Q(16x64) * K(64x32)
    v16h Kb[2][2];
    #pragma unroll
    for (int nt = 0; nt < 2; nt++) {
      const int j = j0 + nt * 16 + l15;
      #pragma unroll
      for (int ch = 0; ch < 2; ch++) {
        const _Float16* kp = kb + (size_t)j * NC8 + ch * 32 + hf * 16;
        Kb[nt][ch] = cat16(*(const v8h*)(kp), *(const v8h*)(kp + 8));
      }
    }
    v8f e0, e1;
    {
      v8f z = {};
      e0 = wmma16(Qm[1], Kb[0][1], z);
      e0 = wmma16(Qm[0], Kb[0][0], e0);
      v8f z2 = {};
      e1 = wmma16(Qm[1], Kb[1][1], z2);
      e1 = wmma16(Qm[0], Kb[1][0], e1);
    }

    // p = exp(e - m): write to per-wave LDS (C layout -> A layout transpose)
    #pragma unroll
    for (int r = 0; r < 8; r++) {
      float p0 = __expf(e0[r] - rowmax[r]);
      float p1 = __expf(e1[r] - rowmax[r]);
      lsum[r] += p0 + p1;
      const int il = r + hf * 8;
      pp[il * 32 + l15]      = (_Float16)p0;
      pp[il * 32 + 16 + l15] = (_Float16)p1;
    }
    // read P back as WMMA A operand (16 rows x 32 j)
    v16h Pa;
    {
      const _Float16* rp = pp + l15 * 32 + hf * 16;
      Pa = cat16(*(const v8h*)(rp), *(const v8h*)(rp + 8));
    }

    // out^T(16 x 256) += P(16x32) * V^T(32x16) per channel tile
    #pragma unroll
    for (int ct = 0; ct < 16; ct++) {
      const int c = c0 + ct * 16 + l15;
      const _Float16* vp = &ldsVcur[c * 32 + hf * 16];
      v16h Vb = cat16(*(const v8h*)(vp), *(const v8h*)(vp + 8));
      acc[ct] = wmma16(Pa, Vb, acc[ct]);
    }
  }

  // finish softmax denominators
  float invl[8];
  #pragma unroll
  for (int r = 0; r < 8; r++) {
    float s = lsum[r];
    s += __shfl_xor(s, 1, 32);
    s += __shfl_xor(s, 2, 32);
    s += __shfl_xor(s, 4, 32);
    s += __shfl_xor(s, 8, 32);
    invl[r] = 1.0f / s;
  }

  const float g = gammap[0];
  float* scr = (float*)pp;   // 16x16 f32 transpose scratch (1 KB, wave-private)

  #pragma unroll
  for (int ct = 0; ct < 16; ct++) {
    #pragma unroll
    for (int r = 0; r < 8; r++) {
      const int il = r + hf * 8;
      scr[il * 16 + l15] = acc[ct][r] * invl[r];
    }
    // wave-private LDS, DS ops are in-order within a wave
    #pragma unroll
    for (int kk = 0; kk < 8; kk++) {
      const int cl = hf + kk * 2;
      float v = scr[l15 * 16 + cl];
      const int cglob = c0 + ct * 16 + cl;
      const int ig    = i0 + mg * 16 + l15;
      const size_t idx = ((size_t)b * NC + cglob) * (size_t)NN + ig;
      out[idx] = g * v + x[idx];
    }
  }
}

// ---------------------------------------------------------------------------
extern "C" void kernel_launch(void* const* d_in, const int* in_sizes, int n_in,
                              void* d_out, int out_size, void* d_ws, size_t ws_size,
                              hipStream_t stream) {
  const float* x     = (const float*)d_in[0];
  const float* wq    = (const float*)d_in[1];
  const float* bq    = (const float*)d_in[2];
  const float* wk    = (const float*)d_in[3];
  const float* bk    = (const float*)d_in[4];
  const float* wv    = (const float*)d_in[5];
  const float* bv    = (const float*)d_in[6];
  const float* gamma = (const float*)d_in[7];
  float* out = (float*)d_out;

  _Float16* qws = (_Float16*)d_ws;                       // [4][4096][64]
  _Float16* kws = qws + (size_t)NB * NN * NC8;           // [4][4096][64]
  _Float16* vws = kws + (size_t)NB * NN * NC8;           // [4][512][4096]

  pam_proj_kernel<<<1280, 256, 0, stream>>>(x, wq, bq, wk, bk, wv, bv,
                                            qws, kws, vws);
  const size_t attn_lds = 65536 + 8192 + 2048;           // 74 KB dynamic LDS
  pam_attn_kernel<<<256, 256, attn_lds, stream>>>(x, gamma, qws, kws, vws, out);
}
